// PonitaFC_44049184587983
// MI455X (gfx1250) — matrix operand order repaired and checked by reading, not compiled, not run on hardware
//
#include <hip/hip_runtime.h>
#include <hip/hip_bf16.h>

typedef __attribute__((ext_vector_type(16))) _Float16 v16h;
typedef __attribute__((ext_vector_type(8)))  float    v8f;

#define NORI 20
#define NPTS 160
#define PI2  6.283185307179586f

union Frag {
    v16h v;
    uint4 u[2];
};

__device__ __forceinline__ float gelu_f(float x) {
    return 0.5f * x * (1.0f + erff(x * 0.7071067811865476f));
}

// async global->LDS copy of 16B per lane (GLOBAL_LOAD_ASYNC_TO_LDS_B128, ASYNCcnt-tracked)
__device__ __forceinline__ void async_ld_b128(const void* gsrc, void* ldst) {
    unsigned l = (unsigned)(unsigned long long)ldst;   // generic->LDS offset = addr[31:0]
    asm volatile("global_load_async_to_lds_b128 %0, %1, off"
                 :: "v"(l), "v"(gsrc) : "memory");
}
#define ASYNC_WAIT(N) asm volatile("s_wait_asynccnt " #N ::: "memory")

// ---------------------------------------------------------------- weight prep
// transpose f32 [K][N] -> f16 [N][K]
__global__ __launch_bounds__(256) void cvt_t_kernel(const float* __restrict__ s,
                                                    _Float16* __restrict__ d, int K, int N) {
    int idx = blockIdx.x * 256 + threadIdx.x;
    if (idx < K * N) {
        int k = idx / N, n = idx % N;
        d[(long)n * K + k] = (_Float16)s[idx];
    }
}

// bw1 (14x128 f32) -> transposed padded f16 [128][32] (K 14..31 zero)
__global__ __launch_bounds__(128) void bw1p_kernel(const float* __restrict__ bw1,
                                                   _Float16* __restrict__ d) {
    int idx = blockIdx.x * 128 + threadIdx.x;   // 4096 total
    int c = idx >> 5, k = idx & 31;
    d[idx] = (k < 14) ? (_Float16)bw1[k * 128 + c] : (_Float16)0.0f;
}

__global__ __launch_bounds__(256) void copy_kernel(const float* __restrict__ s,
                                                   float* __restrict__ d, int n) {
    int i = blockIdx.x * 256 + threadIdx.x;
    if (i < n) d[i] = s[i];
}

// ---------------------------------------------------------------- fkb: [P*O, 128]
__global__ __launch_bounds__(128) void fkb_kernel(const float* __restrict__ fw1,
                                                  const float* __restrict__ fb1,
                                                  const float* __restrict__ fw2,
                                                  const float* __restrict__ fb2,
                                                  float* __restrict__ fkb) {
    int p = blockIdx.x / NORI, o = blockIdx.x % NORI;
    float tp = (float)p * (PI2 / NORI), to = (float)o * (PI2 / NORI);
    float oi = cosf(tp - to);
    float f0 = oi, f1 = oi * oi, f2 = oi * oi * oi;
    int c = threadIdx.x;
    float mid = gelu_f(f0 * fw1[c] + f1 * fw1[128 + c] + f2 * fw1[256 + c] + fb1[c]);
    __shared__ float sm[128];
    sm[c] = mid;
    __syncthreads();
    float acc = fb2[c];
    for (int j = 0; j < 128; j++) acc += sm[j] * fw2[j * 128 + c];
    fkb[(long)blockIdx.x * 128 + c] = gelu_f(acc);
}

// ---------------------------------------------------------------- fk[l][p*20+o][c]
__global__ __launch_bounds__(128) void fk_kernel(const float* __restrict__ fkb,
                                                 const float* __restrict__ fiber_w,
                                                 float* __restrict__ fk) {
    int l = blockIdx.x / 400, po = blockIdx.x % 400, c = threadIdx.x;
    __shared__ float sb[128];
    sb[c] = fkb[(long)po * 128 + c];
    __syncthreads();
    const float* W = fiber_w + (long)l * 128 * 128;
    float acc = 0.f;
    for (int j = 0; j < 128; j++) acc += sb[j] * W[j * 128 + c];
    fk[(long)blockIdx.x * 128 + c] = acc;
}

// ---------------------------------------------------------------- h = x @ emb_w  [n][o][c]
__global__ __launch_bounds__(128) void emb_kernel(const float* __restrict__ x,
                                                  const float* __restrict__ emb_w,
                                                  float* __restrict__ h) {
    int r = blockIdx.x, c = threadIdx.x;
    __shared__ float sx[16];
    if (threadIdx.x < 16) sx[threadIdx.x] = x[(long)r * 16 + threadIdx.x];
    __syncthreads();
    float acc = 0.f;
#pragma unroll
    for (int i = 0; i < 16; i++) acc += sx[i] * emb_w[i * 128 + c];
    h[(long)r * 128 + c] = acc;
}

// ---------------------------------------------------------------- kb: poly14 -> 128 -> 128 (WMMA)
// One wave per 16-row tile (rows = 16 consecutive n for fixed (o,m)). Out: kb_h[o][m][n][b] f16.
__global__ __launch_bounds__(32) void kb_kernel(const float* __restrict__ pos,
                                                const _Float16* __restrict__ bw1p,  // [128][32] T,padded
                                                const float* __restrict__ bb1,
                                                const _Float16* __restrict__ bw2T,  // [128][128] T
                                                const float* __restrict__ bb2,
                                                _Float16* __restrict__ kb_h) {
    int t = blockIdx.x;
    int o = t / 1600;
    int rem = t % 1600;
    int m = rem / 10;
    int n0 = (rem % 10) * 16;
    int l = threadIdx.x;
    int row = l & 15;
    int hi8 = (l & 16) >> 1;   // 0 or 8
    int n = n0 + row;

    float th = (float)o * (PI2 / NORI);
    float ox = cosf(th), oy = sinf(th);
    float rx = pos[n * 2 + 0] - pos[m * 2 + 0];
    float ry = pos[n * 2 + 1] - pos[m * 2 + 1];
    float s0 = rx * ox + ry * oy;                 // inv1
    float ex = rx - rx * s0, ey = ry - ry * s0;
    float s1 = sqrtf(ex * ex + ey * ey);          // inv2
    float f[14] = { s0, s1,
                    s0 * s0, s0 * s1, s1 * s0, s1 * s1,
                    s0 * s0 * s0, s0 * s0 * s1, s0 * s1 * s0, s0 * s1 * s1,
                    s1 * s0 * s0, s1 * s0 * s1, s1 * s1 * s0, s1 * s1 * s1 };

    v16h a1;
#pragma unroll
    for (int i = 0; i < 16; i++) {
        int Kb = (i & 7) + ((i & 8) << 1);        // constant after unroll
        float v0 = (Kb < 14) ? f[Kb] : 0.f;
        float v1 = (Kb + 8 < 14) ? f[Kb + 8] : 0.f;
        a1[i] = (_Float16)(hi8 ? v1 : v0);
    }

    __shared__ __align__(16) _Float16 smid[16 * 128];
    int colb = l & 15;

    // GEMM1: [16x14]x[14x128] + gelu -> smid
#pragma unroll 1
    for (int nt = 0; nt < 8; nt++) {
        int col = nt * 16 + colb;
        Frag b;
        const uint4* bp = (const uint4*)&bw1p[col * 32 + hi8];
        b.u[0] = bp[0];
        b.u[1] = bp[2];
        v8f acc;
        float bias = bb1[col];
#pragma unroll
        for (int v = 0; v < 8; v++) acc[v] = bias;
        acc = __builtin_amdgcn_wmma_f32_16x16x32_f16(false, a1, false, b.v, (short)0, acc, false, false);
#pragma unroll
        for (int v = 0; v < 8; v++) smid[(v + hi8) * 128 + col] = (_Float16)gelu_f(acc[v]);
    }
    __syncthreads();

    // GEMM2: [16x128]x[128x128] + gelu -> kb_h
    long base = (((long)o * NPTS + m) * NPTS + n0) * 128;
#pragma unroll 1
    for (int nt = 0; nt < 8; nt++) {
        int col = nt * 16 + colb;
        v8f acc;
        float bias = bb2[col];
#pragma unroll
        for (int v = 0; v < 8; v++) acc[v] = bias;
#pragma unroll
        for (int kk = 0; kk < 4; kk++) {
            Frag a, b;
            a.u[0] = *(const uint4*)&smid[row * 128 + kk * 32 + hi8];
            a.u[1] = *(const uint4*)&smid[row * 128 + kk * 32 + 16 + hi8];
            const uint4* bp = (const uint4*)&bw2T[col * 128 + kk * 32 + hi8];
            b.u[0] = bp[0];
            b.u[1] = bp[2];
            acc = __builtin_amdgcn_wmma_f32_16x16x32_f16(false, a.v, false, b.v, (short)0, acc, false, false);
        }
#pragma unroll
        for (int v = 0; v < 8; v++)
            kb_h[base + (long)(v + hi8) * 128 + col] = (_Float16)gelu_f(acc[v]);
    }
}

// ---------------------------------------------------------------- spatial conv as GEMM per o:
// x1[m,c] = sum_{n,b} kb[o][m][n][b] * (kw[b][c]*h[n][o][c]).  Output x1[o][m][c].
// kwT staged in LDS; A tiles double-buffered via async global->LDS.
__global__ __launch_bounds__(256) void conv_kernel(const _Float16* __restrict__ kb_h,
                                                   const float* __restrict__ h,
                                                   const _Float16* __restrict__ kwT, // [c][b] T
                                                   float* __restrict__ x1) {
    int o = blockIdx.x / 10, mt = blockIdx.x % 10;
    int tid = threadIdx.x;
    int wave = tid >> 5, l = tid & 31;
    int row = l & 15, hi8 = (l & 16) >> 1;
    int col = wave * 16 + (l & 15);

    __shared__ __align__(16) _Float16 sW[128 * 128];   // 32 KB: kernel_w[l] transposed
    __shared__ __align__(16) _Float16 sA[2][16 * 128]; // 2 x 4 KB double buffer

#pragma unroll
    for (int q = 0; q < 8; q++) {                      // 16384 f16 cooperative load
        int e = (q * 256 + tid) * 8;
        *(uint4*)&sW[e] = *(const uint4*)&kwT[e];
    }

    v8f acc;
#pragma unroll
    for (int v = 0; v < 8; v++) acc[v] = 0.f;

    int lm = tid >> 4;            // 0..15: A-tile row
    int lc = (tid & 15) * 8;      // 8-f16 chunk
    const long oslab = (long)o * NPTS * NPTS * 128;
    const _Float16* abase = kb_h + oslab + ((long)(mt * 16 + lm) * NPTS) * 128 + lc;

    async_ld_b128(abase, &sA[0][lm * 128 + lc]);       // prologue: tile n=0
    __syncthreads();                                   // sW ready (also orders buffers)

    for (int n = 0; n < NPTS; n++) {
        int cur = n & 1;
        if (n + 1 < NPTS) {
            async_ld_b128(abase + (long)(n + 1) * 128, &sA[cur ^ 1][lm * 128 + lc]);
            ASYNC_WAIT(0x1);                           // tile n landed (in-order completion)
        } else {
            ASYNC_WAIT(0x0);
        }
        __syncthreads();

        float hv = h[((long)n * NORI + o) * 128 + col];
        _Float16 hh = (_Float16)hv;
        v16h hs;
#pragma unroll
        for (int i = 0; i < 16; i++) hs[i] = hh;

#pragma unroll
        for (int bb = 0; bb < 4; bb++) {
            int b0 = bb * 32;
            Frag a, w;
            a.u[0] = *(const uint4*)&sA[cur][row * 128 + b0 + hi8];
            a.u[1] = *(const uint4*)&sA[cur][row * 128 + b0 + 16 + hi8];
            int wb = col * 128 + b0 + hi8;
            w.u[0] = *(const uint4*)&sW[wb];
            w.u[1] = *(const uint4*)&sW[wb + 16];
            v16h bv = w.v * hs;                        // packed f16 scale by h
            acc = __builtin_amdgcn_wmma_f32_16x16x32_f16(false, a.v, false, bv, (short)0, acc, false, false);
        }
        __syncthreads();                               // done reading sA[cur] before overwrite
    }
#pragma unroll
    for (int v = 0; v < 8; v++)
        x1[((long)o * NPTS + mt * 16 + v + hi8) * 128 + col] = acc[v];
}

// ---------------------------------------------------------------- fiber conv + /C + bias + LayerNorm
__global__ __launch_bounds__(128) void fiber_ln_kernel(const float* __restrict__ x1,
                                                       const float* __restrict__ fk_l,
                                                       const float* __restrict__ conv_b_l,
                                                       const float* __restrict__ ln_g_l,
                                                       const float* __restrict__ ln_b_l,
                                                       _Float16* __restrict__ xnh) {
    int m = blockIdx.x / NORI, p = blockIdx.x % NORI, c = threadIdx.x;
    float acc = 0.f;
    for (int o = 0; o < NORI; o++)
        acc += x1[((long)o * NPTS + m) * 128 + c] * fk_l[(p * NORI + o) * 128 + c];
    float x2 = acc * (1.0f / 128.0f) + conv_b_l[c];
    __shared__ float rs[128], rq[128];
    rs[c] = x2;
    rq[c] = x2 * x2;
    __syncthreads();
    for (int s = 64; s > 0; s >>= 1) {
        if (c < s) { rs[c] += rs[c + s]; rq[c] += rq[c + s]; }
        __syncthreads();
    }
    float mu  = rs[0] * (1.f / 128.f);
    float var = rq[0] * (1.f / 128.f) - mu * mu;
    float xn = (x2 - mu) * rsqrtf(var + 1e-5f) * ln_g_l[c] + ln_b_l[c];
    xnh[(long)blockIdx.x * 128 + c] = (_Float16)xn;
}

// ---------------------------------------------------------------- MLP1: [3200x128]x[128x512] + gelu
__global__ __launch_bounds__(256) void mlp1_kernel(const _Float16* __restrict__ xnh,
                                                   const _Float16* __restrict__ w1T, // [512][128] T
                                                   const float* __restrict__ b1,
                                                   _Float16* __restrict__ midh) {
    int rt = blockIdx.x >> 2, cg = blockIdx.x & 3;
    int tid = threadIdx.x;
    int wave = tid >> 5, l = tid & 31;
    int row = l & 15, hi8 = (l & 16) >> 1;
    int col = cg * 128 + wave * 16 + (l & 15);
    __shared__ __align__(16) _Float16 sA[16 * 128];
    int lm = tid >> 4, lc = (tid & 15) * 8;
    *(uint4*)&sA[lm * 128 + lc] = *(const uint4*)&xnh[((long)rt * 16 + lm) * 128 + lc];
    __syncthreads();
    v8f acc;
    float bias = b1[col];
#pragma unroll
    for (int v = 0; v < 8; v++) acc[v] = bias;
#pragma unroll
    for (int kk = 0; kk < 4; kk++) {
        Frag a, b;
        a.u[0] = *(const uint4*)&sA[row * 128 + kk * 32 + hi8];
        a.u[1] = *(const uint4*)&sA[row * 128 + kk * 32 + 16 + hi8];
        const uint4* bp = (const uint4*)&w1T[(long)col * 128 + kk * 32 + hi8];
        b.u[0] = bp[0];
        b.u[1] = bp[2];
        acc = __builtin_amdgcn_wmma_f32_16x16x32_f16(false, a.v, false, b.v, (short)0, acc, false, false);
    }
#pragma unroll
    for (int v = 0; v < 8; v++)
        midh[((long)rt * 16 + v + hi8) * 512 + col] = (_Float16)gelu_f(acc[v]);
}

// ---------------------------------------------------------------- MLP2: [3200x512]x[512x128] + residual
__global__ __launch_bounds__(256) void mlp2_kernel(const _Float16* __restrict__ midh,
                                                   const _Float16* __restrict__ w2T, // [128][512] T
                                                   const float* __restrict__ b2,
                                                   float* __restrict__ h) {
    int rt = blockIdx.x;
    int tid = threadIdx.x;
    int wave = tid >> 5, l = tid & 31;
    int row = l & 15, hi8 = (l & 16) >> 1;
    int col = wave * 16 + (l & 15);
    __shared__ __align__(16) _Float16 sA[16 * 512];
#pragma unroll
    for (int q = 0; q < 4; q++) {
        int e = (q * 256 + tid) * 8;
        *(uint4*)&sA[e] = *(const uint4*)&midh[(long)rt * 16 * 512 + e];
    }
    __syncthreads();
    v8f acc;
    float bias = b2[col];
#pragma unroll
    for (int v = 0; v < 8; v++) acc[v] = bias;
#pragma unroll 4
    for (int kk = 0; kk < 16; kk++) {
        Frag a, b;
        a.u[0] = *(const uint4*)&sA[row * 512 + kk * 32 + hi8];
        a.u[1] = *(const uint4*)&sA[row * 512 + kk * 32 + 16 + hi8];
        const uint4* bp = (const uint4*)&w2T[(long)col * 512 + kk * 32 + hi8];
        b.u[0] = bp[0];
        b.u[1] = bp[2];
        acc = __builtin_amdgcn_wmma_f32_16x16x32_f16(false, a.v, false, b.v, (short)0, acc, false, false);
    }
#pragma unroll
    for (int v = 0; v < 8; v++) {
        long idx = ((long)rt * 16 + v + hi8) * 128 + col;
        h[idx] = h[idx] + acc[v];
    }
}

// ================================================================ launcher
extern "C" void kernel_launch(void* const* d_in, const int* in_sizes, int n_in,
                              void* d_out, int out_size, void* d_ws, size_t ws_size,
                              hipStream_t stream) {
    const float* x        = (const float*)d_in[0];
    const float* pos      = (const float*)d_in[1];
    const float* bw1      = (const float*)d_in[2];
    const float* bb1      = (const float*)d_in[3];
    const float* bw2      = (const float*)d_in[4];
    const float* bb2      = (const float*)d_in[5];
    const float* fw1      = (const float*)d_in[6];
    const float* fb1      = (const float*)d_in[7];
    const float* fw2      = (const float*)d_in[8];
    const float* fb2      = (const float*)d_in[9];
    const float* emb_w    = (const float*)d_in[10];
    const float* kernel_w = (const float*)d_in[11];
    const float* fiber_w  = (const float*)d_in[12];
    const float* conv_b   = (const float*)d_in[13];
    const float* ln_g     = (const float*)d_in[14];
    const float* ln_b     = (const float*)d_in[15];
    const float* mlp_w1   = (const float*)d_in[16];
    const float* mlp_b1   = (const float*)d_in[17];
    const float* mlp_w2   = (const float*)d_in[18];
    const float* mlp_b2   = (const float*)d_in[19];
    (void)in_sizes; (void)n_in; (void)ws_size;

    char* w = (char*)d_ws;
    auto carve = [&](size_t bytes) {
        void* p = (void*)w;
        w += (bytes + 255) & ~(size_t)255;
        return p;
    };
    _Float16* kb_h  = (_Float16*)carve((size_t)NORI * NPTS * NPTS * 128 * 2); // 131 MB
    float*    fkb   = (float*)carve(400 * 128 * 4);
    float*    fk    = (float*)carve(2 * 400 * 128 * 4);
    float*    h     = (float*)carve((size_t)NPTS * NORI * 128 * 4);
    float*    x1    = (float*)carve((size_t)NORI * NPTS * 128 * 4);
    _Float16* xnh   = (_Float16*)carve((size_t)NPTS * NORI * 128 * 2);
    _Float16* midh  = (_Float16*)carve((size_t)NPTS * NORI * 512 * 2);
    _Float16* bw1p  = (_Float16*)carve(128 * 32 * 2);
    _Float16* bw2T  = (_Float16*)carve(128 * 128 * 2);
    _Float16* kwT   = (_Float16*)carve(2 * 128 * 128 * 2);
    _Float16* w1T   = (_Float16*)carve(2 * 512 * 128 * 2);
    _Float16* w2T   = (_Float16*)carve(2 * 128 * 512 * 2);

    // ---- weight prep (f32 -> transposed f16)
    bw1p_kernel<<<32, 128, 0, stream>>>(bw1, bw1p);
    cvt_t_kernel<<<64, 256, 0, stream>>>(bw2, bw2T, 128, 128);
    for (int l = 0; l < 2; l++) {
        cvt_t_kernel<<<64, 256, 0, stream>>>(kernel_w + (size_t)l * 16384, kwT + (size_t)l * 16384, 128, 128);
        cvt_t_kernel<<<256, 256, 0, stream>>>(mlp_w1 + (size_t)l * 65536, w1T + (size_t)l * 65536, 128, 512);
        cvt_t_kernel<<<256, 256, 0, stream>>>(mlp_w2 + (size_t)l * 65536, w2T + (size_t)l * 65536, 512, 128);
    }

    // ---- fiber basis, fiber kernels, embedding
    fkb_kernel<<<400, 128, 0, stream>>>(fw1, fb1, fw2, fb2, fkb);
    fk_kernel<<<800, 128, 0, stream>>>(fkb, fiber_w, fk);
    emb_kernel<<<NPTS * NORI, 128, 0, stream>>>(x, emb_w, h);

    // ---- spatial kernel basis kb (big WMMA stage)
    kb_kernel<<<NORI * NPTS * 10, 32, 0, stream>>>(pos, bw1p, bb1, bw2T, bb2, kb_h);

    // ---- layers
    for (int l = 0; l < 2; l++) {
        conv_kernel<<<NORI * 10, 256, 0, stream>>>(kb_h, h, kwT + (size_t)l * 16384, x1);
        fiber_ln_kernel<<<NPTS * NORI, 128, 0, stream>>>(x1, fk + (size_t)l * 400 * 128,
                                                         conv_b + l * 128, ln_g + l * 128,
                                                         ln_b + l * 128, xnh);
        mlp1_kernel<<<200 * 4, 256, 0, stream>>>(xnh, w1T + (size_t)l * 65536,
                                                 mlp_b1 + l * 512, midh);
        mlp2_kernel<<<200, 256, 0, stream>>>(midh, w2T + (size_t)l * 65536,
                                             mlp_b2 + l * 128, h);
    }

    // ---- output
    copy_kernel<<<(out_size + 255) / 256, 256, 0, stream>>>(h, (float*)d_out, out_size);
}
